// GatedFFN_2465311228262
// MI455X (gfx1250) — matrix operand order repaired and checked by reading, not compiled, not run on hardware
//
#include <hip/hip_runtime.h>

#define DEV static __device__ __forceinline__

constexpr int D_MODEL = 1024;
constexpr int D_FF    = 4096;
constexpr int NEXP    = 8;
constexpr int TM      = 32;            // token tile (M): 2 x 16 m-subtiles
constexpr int CHUNK   = 512;           // D_FF chunk resident in LDS
constexpr int NCHUNK  = D_FF / CHUNK;  // 8

typedef __attribute__((ext_vector_type(16))) __bf16 v16bf;
typedef __attribute__((ext_vector_type(8)))  float  v8f;

DEV v8f wmma_bf16(v16bf a, v16bf b, v8f c) {
  // D = A(16x32 bf16) * B(32x16 bf16) + C(16x16 f32)
  return __builtin_amdgcn_wmma_f32_16x16x32_bf16(false, a, false, b, (short)0, c,
                                                 false, false);
}

// A fragment (16x32) per ISA 7.12.2: lane L -> M = L&15, half = L>>4.
// VGPR0-3 hold K = k0 + half*8 + {0..7}; VGPR4-7 hold K = k0 + 16 + half*8 + {0..7}.
DEV v16bf a_frag_f32(const float* base, int ld, int k0, int lane) {
  const float* lo = base + (size_t)(lane & 15) * ld + k0 + ((lane >> 4) * 8);
  const float* hi = lo + 16;
  v16bf a;
#pragma unroll
  for (int i = 0; i < 8; ++i) { a[i] = (__bf16)lo[i]; a[8 + i] = (__bf16)hi[i]; }
  return a;
}

DEV v16bf a_frag_bf16(const __bf16* base, int ld, int k0, int lane) {
  const __bf16* lo = base + (size_t)(lane & 15) * ld + k0 + ((lane >> 4) * 8);
  const __bf16* hi = lo + 16;
  v16bf a;
#pragma unroll
  for (int i = 0; i < 8; ++i) { a[i] = lo[i]; a[8 + i] = hi[i]; }
  return a;
}

// B fragment (32x16): lane L holds column N = n0 + (L&15); elements are K in table order
// (K = k0 + half*8 + {0..7}, then k0 + 16 + half*8 + {0..7}) (f32 -> bf16).
// p must already point at &W[k0 + half*8][n0 + (lane&15)] in a row-major [K][N] matrix.
DEV v16bf b_frag_f32(const float* p, int ld) {
  v16bf b;
#pragma unroll
  for (int i = 0; i < 16; ++i) b[i] = (__bf16)p[(size_t)(i < 8 ? i : i + 8) * ld];
  return b;
}

// ---------------------------------------------------------------------------
// Kernel 1: top-1 gate + one-hot output + bucket tokens per expert.
// One wave32 per token; 4 tokens per 128-thread block.
// ---------------------------------------------------------------------------
__global__ void __launch_bounds__(128)
gate_top1_kernel(const float* __restrict__ x, const float* __restrict__ gW,
                 const float* __restrict__ gb, float* __restrict__ gate_out,
                 int* __restrict__ counts, int* __restrict__ sidx, int cap) {
  const int lane = threadIdx.x & 31;
  const int wav  = threadIdx.x >> 5;
  const int tok  = blockIdx.x * 4 + wav;
  const float* xr = x + (size_t)tok * D_MODEL;

  float acc[NEXP];
#pragma unroll
  for (int e = 0; e < NEXP; ++e) acc[e] = 0.f;

  for (int i = 0; i < D_MODEL / 32; ++i) {
    const int k = i * 32 + lane;
    const float xv = xr[k];
    const float* w = gW + (size_t)k * NEXP;
#pragma unroll
    for (int e = 0; e < NEXP; ++e) acc[e] += xv * w[e];
  }
#pragma unroll
  for (int off = 16; off > 0; off >>= 1) {
#pragma unroll
    for (int e = 0; e < NEXP; ++e) acc[e] += __shfl_xor(acc[e], off, 32);
  }
  if (lane == 0) {
    int best = 0;
    float bv = acc[0] + gb[0];
#pragma unroll
    for (int e = 1; e < NEXP; ++e) {
      const float v = acc[e] + gb[e];
      if (v > bv) { bv = v; best = e; }   // strict > == first-max (jnp.argmax)
    }
#pragma unroll
    for (int e = 0; e < NEXP; ++e)
      gate_out[(size_t)tok * NEXP + e] = (e == best) ? 1.f : 0.f;
    const int pos = atomicAdd(&counts[best], 1);
    sidx[best * cap + pos] = tok;
  }
}

// ---------------------------------------------------------------------------
// Kernel 2: fused per-expert FFN.  Block = 32 tokens of one expert, 8 waves.
//   LDS: xs [32][1024] f32 (128KB, async-gathered) | hs [32][512] bf16 (32KB)
//   Each B fragment (weights) feeds 2 WMMAs (two 16-row m-subtiles), halving
//   the dominant weight-stream traffic.  h never touches HBM.
// ---------------------------------------------------------------------------
__global__ void __launch_bounds__(256)
moe_ffn_kernel(const float* __restrict__ x,
               const float* __restrict__ upW, const float* __restrict__ upB,
               const float* __restrict__ dnW, const float* __restrict__ dnB,
               const int* __restrict__ counts, const int* __restrict__ sidx,
               float* __restrict__ out, int cap) {
  const int e    = blockIdx.y;
  const int cnt  = counts[e];
  const int row0 = blockIdx.x * TM;
  if (row0 >= cnt) return;                 // wave-uniform: EXEC stays full

  extern __shared__ char smem[];
  float*  xs   = (float*)smem;                        // 32 x 1024 f32 (128KB)
  __bf16* hs   = (__bf16*)(smem + 128 * 1024);        // 32 x 512 bf16  (32KB)
  int*    rows = (int*)(smem + 160 * 1024);           // 32 token ids (-1 = pad)

  const int tid  = threadIdx.x;
  const int lane = tid & 31;
  const int wav  = tid >> 5;               // 0..7
  const int half = lane >> 4;
  const int nl   = lane & 15;

  if (tid < TM) {
    const int r = row0 + tid;
    rows[tid] = (r < cnt) ? sidx[e * cap + r] : -1;
  }
  __syncthreads();

  // Gather 32 (scattered) token rows of x straight into LDS with async copy
  // (GLOBAL_LOAD_ASYNC_TO_LDS_B128, tracked on ASYNCcnt); zero-fill pad rows.
  for (int j = tid; j < TM * (D_MODEL / 4); j += 256) {
    const int m = j >> 8;                  // D_MODEL/4 = 256 float4 per row
    const int c = (j & 255) * 4;
    const int t = rows[m];
    float* dst = xs + (size_t)m * D_MODEL + c;
    if (t >= 0) {
      const float* src = x + (size_t)t * D_MODEL + c;
      const unsigned lds = (unsigned)(uintptr_t)dst;   // flat->LDS = addr[31:0]
      asm volatile("global_load_async_to_lds_b128 %0, %1, off"
                   :: "v"(lds), "v"(src) : "memory");
    } else {
      *(float4*)dst = make_float4(0.f, 0.f, 0.f, 0.f);
    }
  }
  asm volatile("s_wait_asynccnt 0x0" ::: "memory");
  __syncthreads();

  const float* upWe = upW + (size_t)e * D_MODEL * D_FF;
  const float* dnWe = dnW + (size_t)e * D_FF * D_MODEL;

  v8f acc0[8], acc1[8];                    // 32x128 down-output per wave
#pragma unroll
  for (int j = 0; j < 8; ++j) {
    acc0[j] = (v8f){0.f, 0.f, 0.f, 0.f, 0.f, 0.f, 0.f, 0.f};
    acc1[j] = (v8f){0.f, 0.f, 0.f, 0.f, 0.f, 0.f, 0.f, 0.f};
  }

  for (int c = 0; c < NCHUNK; ++c) {
    const int ff0 = c * CHUNK;

    // ---- up projection: this wave computes h columns [wav*64, wav*64+64)
    for (int j = 0; j < 4; ++j) {
      const int ncol = wav * 64 + j * 16;            // chunk-local N
      v8f h0 = (v8f){0.f, 0.f, 0.f, 0.f, 0.f, 0.f, 0.f, 0.f};
      v8f h1 = (v8f){0.f, 0.f, 0.f, 0.f, 0.f, 0.f, 0.f, 0.f};
      for (int ks = 0; ks < D_MODEL / 32; ++ks) {
        const int k0 = ks * 32;
        const v16bf a0 = a_frag_f32(xs, D_MODEL, k0, lane);
        const v16bf a1 = a_frag_f32(xs + 16 * D_MODEL, D_MODEL, k0, lane);
        const float* wp = upWe + (size_t)(k0 + half * 8) * D_FF + ff0 + ncol + nl;
        if (ks + 1 < D_MODEL / 32)
          __builtin_prefetch(wp + (size_t)32 * D_FF, 0, 0);  // next K rows
        const v16bf b = b_frag_f32(wp, D_FF);    // one B feeds both m-subtiles
        h0 = wmma_bf16(a0, b, h0);
        h1 = wmma_bf16(a1, b, h1);
      }
      const float bias = upB[(size_t)e * D_FF + ff0 + ncol + nl];
#pragma unroll
      for (int v = 0; v < 8; ++v) {        // C layout: M = v + 8*half, N = nl
        float h0v = h0[v] + bias;
        float h1v = h1[v] + bias;
        h0v = h0v > 0.f ? h0v : 0.f;       // ReLU
        h1v = h1v > 0.f ? h1v : 0.f;
        hs[(size_t)(v + 8 * half) * CHUNK + ncol + nl]        = (__bf16)h0v;
        hs[(size_t)(16 + v + 8 * half) * CHUNK + ncol + nl]   = (__bf16)h1v;
      }
    }
    __syncthreads();

    // ---- down projection: acc += h_chunk @ down_W[ff0:ff0+CHUNK, :]
    for (int ks = 0; ks < CHUNK / 32; ++ks) {
      const int k0 = ks * 32;
      const v16bf a0 = a_frag_bf16(hs, CHUNK, k0, lane);
      const v16bf a1 = a_frag_bf16(hs + 16 * CHUNK, CHUNK, k0, lane);
      const float* wrow = dnWe + (size_t)(ff0 + k0 + half * 8) * D_MODEL + wav * 128 + nl;
      if (ks + 1 < CHUNK / 32)
        __builtin_prefetch(wrow + (size_t)32 * D_MODEL, 0, 0);
#pragma unroll
      for (int j = 0; j < 8; ++j) {
        const v16bf b = b_frag_f32(wrow + j * 16, D_MODEL);
        acc0[j] = wmma_bf16(a0, b, acc0[j]);
        acc1[j] = wmma_bf16(a1, b, acc1[j]);
      }
    }
    __syncthreads();                       // before next chunk rewrites hs
  }

  // Epilogue: down bias, scatter rows back to original token positions.
#pragma unroll
  for (int j = 0; j < 8; ++j) {
    const int n = wav * 128 + j * 16 + nl;
    const float bias = dnB[(size_t)e * D_MODEL + n];
#pragma unroll
    for (int v = 0; v < 8; ++v) {
      const int m = v + 8 * half;
      const int t0 = rows[m];
      const int t1 = rows[16 + m];
      if (t0 >= 0) out[(size_t)t0 * D_MODEL + n] = acc0[j][v] + bias;
      if (t1 >= 0) out[(size_t)t1 * D_MODEL + n] = acc1[j][v] + bias;
    }
  }
}

extern "C" void kernel_launch(void* const* d_in, const int* in_sizes, int n_in,
                              void* d_out, int out_size, void* d_ws, size_t ws_size,
                              hipStream_t stream) {
  const float* x   = (const float*)d_in[0];
  const float* gW  = (const float*)d_in[1];
  const float* gb  = (const float*)d_in[2];
  const float* upW = (const float*)d_in[3];
  const float* upB = (const float*)d_in[4];
  const float* dnW = (const float*)d_in[5];
  const float* dnB = (const float*)d_in[6];

  const int ntok = in_sizes[0] / D_MODEL;           // B*T = 8192
  float* out  = (float*)d_out;                      // [ntok * 1024]
  float* gate = out + (size_t)ntok * D_MODEL;       // [ntok * 8]

  int* counts = (int*)d_ws;                         // 16 ints (padded)
  int* sidx   = counts + 16;                        // NEXP * ntok ints
  hipMemsetAsync(counts, 0, 16 * sizeof(int), stream);

  gate_top1_kernel<<<ntok / 4, 128, 0, stream>>>(x, gW, gb, gate, counts, sidx,
                                                 ntok);

  const size_t smem = 160 * 1024 + 256;             // xs + hs + rows
  moe_ffn_kernel<<<dim3(ntok / TM, NEXP), 256, smem, stream>>>(
      x, upW, upB, dnW, dnB, counts, sidx, out, ntok);
}
